// PointNet_56324201119982
// MI455X (gfx1250) — matrix operand order repaired and checked by reading, not compiled, not run on hardware
//
#include <hip/hip_runtime.h>
#include <hip/hip_bf16.h>
#include <math.h>

typedef __attribute__((ext_vector_type(16))) _Float16 v16h;
typedef __attribute__((ext_vector_type(8)))  _Float16 v8h;
typedef __attribute__((ext_vector_type(8)))  float    v8f;

union AFrag { v16h v; v8h h[2]; _Float16 e[16]; };

__device__ __forceinline__ v8f wmma16(v16h a, v16h b, v8f c) {
  // D = A(16x32 f16) * B(32x16 f16) + C(16x16 f32)
  return __builtin_amdgcn_wmma_f32_16x16x32_f16(false, a, false, b, (short)0, c,
                                                false, false);
}

// B fragment stored lane-major: frag base + lane*16 halves (32B per lane).
__device__ __forceinline__ v16h load_bfrag(const _Float16* fp, int lane) {
  AFrag u;
  const v8h* q = (const v8h*)(fp + lane * 16);
  u.h[0] = q[0];
  u.h[1] = q[1];
  return u.v;
}

// A fragment from a row-major f16 LDS tile (16 rows x width cols), K-chunk base kb.
// Lanes 0-15 hold K = kb+[0..7],[16..23]; lanes 16-31 hold K = kb+[8..15],[24..31].
__device__ __forceinline__ v16h lds_afrag(const _Float16* tile, int width, int kb, int lane) {
  const int m = lane & 15;
  const int off = (lane & 16) ? 8 : 0;
  const _Float16* p = tile + m * width + kb + off;
  AFrag u;
  u.h[0] = *(const v8h*)(p);
  u.h[1] = *(const v8h*)(p + 16);
  return u.v;
}

__device__ __forceinline__ void lds_fence() {
  asm volatile("s_wait_dscnt 0" ::: "memory");
}

// Order-preserving float max via integer atomics (values init'd to -inf).
__device__ __forceinline__ void atomic_fmax(float* addr, float v) {
  if (v >= 0.0f) atomicMax((int*)addr, __float_as_int(v));
  else           atomicMin((unsigned int*)addr, __float_as_uint(v));
}

__global__ void init_agg(float* __restrict__ agg, int n) {
  for (int i = blockIdx.x * blockDim.x + threadIdx.x; i < n;
       i += gridDim.x * blockDim.x)
    agg[i] = __int_as_float(0xff800000); // -inf
}

// Repack row-major f32 weights [K x Nw] into WMMA B fragments (f16),
// frag index = kc*ntCount + nt, each fragment = 512 halves, lane-major.
__global__ void pack_b(const float* __restrict__ W, _Float16* __restrict__ dst,
                       int K, int Nw, int kcCount, int ntCount) {
  const int total = kcCount * ntCount * 512;
  for (int i = blockIdx.x * blockDim.x + threadIdx.x; i < total;
       i += gridDim.x * blockDim.x) {
    const int frag = i >> 9;
    const int w    = i & 511;
    const int lane = w >> 4;
    const int h    = w & 15;
    const int kc   = frag / ntCount;
    const int nt   = frag - kc * ntCount;
    const int k    = kc * 32 + ((lane & 16) ? 8 : 0) + (h < 8 ? h : h + 8);
    const int n    = nt * 16 + (lane & 15);
    const float v  = (k < K && n < Nw) ? W[k * Nw + n] : 0.0f;
    dst[i] = (_Float16)v;
  }
}

// ---------------- Edge phase: fused message MLP + segment-max ----------------
__global__ void __launch_bounds__(256) edge_msg_agg(
    const float* __restrict__ x, const float* __restrict__ pos,
    const long long* __restrict__ ei,
    const float* __restrict__ b1, const float* __restrict__ b2,
    const _Float16* __restrict__ w1p, const _Float16* __restrict__ w2p,
    float* __restrict__ agg, int E_, int N_, int numTiles) {
  __shared__ _Float16 smem[8][16 * 64];
  const int lane = threadIdx.x & 31;
  const int wave = threadIdx.x >> 5;
  _Float16* tile = &smem[wave][0];
  const int col  = lane & 15;
  const int mOff = (lane & 16) ? 8 : 0;

  // Resident weight fragments (tiny: 6x64 and 64x64).
  v16h w1f[4];
#pragma unroll
  for (int nt = 0; nt < 4; ++nt) w1f[nt] = load_bfrag(w1p + nt * 512, lane);
  v16h w2f[2][4];
#pragma unroll
  for (int kc = 0; kc < 2; ++kc)
#pragma unroll
    for (int nt = 0; nt < 4; ++nt)
      w2f[kc][nt] = load_bfrag(w2p + (kc * 4 + nt) * 512, lane);

  const int gw = blockIdx.x * (blockDim.x >> 5) + wave;
  const int nw = gridDim.x * (blockDim.x >> 5);
  const int R  = E_ + N_;

  for (int t = gw; t < numTiles; t += nw) {
    const int r = t * 16 + col;
    int src, dst;
    const int valid = (r < R) ? 1 : 0;
    if (r < E_) { src = (int)ei[r]; dst = (int)ei[E_ + r]; }
    else        { int k = r - E_; if (k >= N_) k = 0; src = k; dst = k; }

    // Prefetch next tile's edge indices for this wave (global_prefetch_b8).
    {
      const int rn = (t + nw) * 16 + col;
      if (rn < E_) {
        __builtin_prefetch(ei + rn, 0, 0);
        __builtin_prefetch(ei + E_ + rn, 0, 0);
      }
    }

    // Build A fragment for msg = [x_j, pos_j - pos_i] (K=6, zero padded to 32).
    AFrag a1;
#pragma unroll
    for (int j = 0; j < 16; ++j) a1.e[j] = (_Float16)0.0f;
    if (lane < 16) { // lanes 0-15 carry K=0..7; K>=6 stays zero
      a1.e[0] = (_Float16)x[src * 3 + 0];
      a1.e[1] = (_Float16)x[src * 3 + 1];
      a1.e[2] = (_Float16)x[src * 3 + 2];
      a1.e[3] = (_Float16)(pos[src * 3 + 0] - pos[dst * 3 + 0]);
      a1.e[4] = (_Float16)(pos[src * 3 + 1] - pos[dst * 3 + 1]);
      a1.e[5] = (_Float16)(pos[src * 3 + 2] - pos[dst * 3 + 2]);
    }

    // Layer 1: relu(msg@W1+b1) -> 16x64 f16 tile in LDS (A-repack staging).
#pragma unroll
    for (int nt = 0; nt < 4; ++nt) {
      v8f acc = {};
      acc = wmma16(a1.v, w1f[nt], acc);
      const float bn = b1[nt * 16 + col];
#pragma unroll
      for (int rr = 0; rr < 8; ++rr) {
        float v = acc[rr] + bn;
        v = v > 0.0f ? v : 0.0f;
        tile[(rr + mOff) * 64 + nt * 16 + col] = (_Float16)v;
      }
    }
    lds_fence();

    const v16h a2_0 = lds_afrag(tile, 64, 0, lane);
    const v16h a2_1 = lds_afrag(tile, 64, 32, lane);

    // Layer 2: h@W2+b2, then float segment-max via atomics.
#pragma unroll
    for (int nt = 0; nt < 4; ++nt) {
      v8f acc = {};
      acc = wmma16(a2_0, w2f[0][nt], acc);
      acc = wmma16(a2_1, w2f[1][nt], acc);
      const float bn = b2[nt * 16 + col];
#pragma unroll
      for (int rr = 0; rr < 8; ++rr) {
        const int m  = rr + mOff;            // row within tile
        const int d  = __shfl(dst, m, 32);   // dst node of that row
        const int vv = __shfl(valid, m, 32);
        if (vv) atomic_fmax(&agg[(size_t)d * 64 + nt * 16 + col], acc[rr] + bn);
      }
    }
  }
}

// ---------------- Node phase: fused global MLP + fc + log_softmax ----------------
// 4 waves/block process 4 tiles in lockstep; W4/W5 chunk fragments are staged
// once per block into LDS and shared by all waves (4x less weight traffic).
__global__ void __launch_bounds__(128) node_mlp(
    const float* __restrict__ agg,
    const float* __restrict__ b3, const float* __restrict__ b4,
    const float* __restrict__ b5, const float* __restrict__ bf,
    const _Float16* __restrict__ w3p, const _Float16* __restrict__ w4p,
    const _Float16* __restrict__ w5p, const _Float16* __restrict__ wfp,
    float* __restrict__ out, int N_, int numTiles) {
  __shared__ _Float16 s_g1[4][16 * 128];     // per-wave relu(agg@W3+b3)
  __shared__ _Float16 s_stage[4][16 * 64];   // per-wave A-repack staging
  __shared__ float    s_out[4][16 * 48];     // per-wave fc output tile
  __shared__ _Float16 s_w4[16 * 512];        // shared W4 chunk fragments (16KB)
  __shared__ _Float16 s_w5[8 * 512];         // shared W5 chunk fragments (8KB)

  const int tid  = threadIdx.x;
  const int lane = tid & 31;
  const int wave = tid >> 5;
  const int col  = lane & 15;
  const int mOff = (lane & 16) ? 8 : 0;
  _Float16* g1    = &s_g1[wave][0];
  _Float16* stage = &s_stage[wave][0];
  float*    outT  = &s_out[wave][0];

  const int tileBlocks = (numTiles + 3) >> 2;

  for (int bt = blockIdx.x; bt < tileBlocks; bt += gridDim.x) {
    const int t = bt * 4 + wave;
    const bool tvalid = (t < numTiles);
    const int base = t * 16;
    int row = base + col;
    if (!tvalid || row >= N_) row = 0;

    // A fragments of agg (16x64 f32 -> f16), loaded directly in A layout.
    v16h a3[2];
#pragma unroll
    for (int kc = 0; kc < 2; ++kc) {
      const float* p = agg + (size_t)row * 64 + kc * 32 + mOff;
      AFrag u;
#pragma unroll
      for (int j = 0; j < 8; ++j) u.e[j]     = (_Float16)p[j];
#pragma unroll
      for (int j = 0; j < 8; ++j) u.e[8 + j] = (_Float16)p[16 + j];
      a3[kc] = u.v;
    }

    // Layer 3: g1 = relu(agg@W3+b3)  (16x128) -> per-wave LDS
#pragma unroll
    for (int nt = 0; nt < 8; ++nt) {
      v8f acc = {};
      acc = wmma16(a3[0], load_bfrag(w3p + (0 * 8 + nt) * 512, lane), acc);
      acc = wmma16(a3[1], load_bfrag(w3p + (1 * 8 + nt) * 512, lane), acc);
      const float bn = b3[nt * 16 + col];
#pragma unroll
      for (int rr = 0; rr < 8; ++rr) {
        float v = acc[rr] + bn;
        v = v > 0.0f ? v : 0.0f;
        g1[(rr + mOff) * 128 + nt * 16 + col] = (_Float16)v;
      }
    }
    lds_fence();

    v16h a4[4];
#pragma unroll
    for (int kc = 0; kc < 4; ++kc) a4[kc] = lds_afrag(g1, 128, kc * 32, lane);

    // Layers 4+5 fused over 16 chunks of 64 hidden cols; weights staged in LDS.
    v8f g3[4];
#pragma unroll
    for (int nt = 0; nt < 4; ++nt) g3[nt] = (v8f){};

    for (int ch = 0; ch < 16; ++ch) {
      __syncthreads();  // WAR: previous chunk's weight reads complete
      // Cooperative stage of this chunk's W4 (16 frags) + W5 (8 frags).
      for (int i = tid; i < 1024; i += 128) {     // 16KB = 1024 * uint4
        const int kc = i >> 8;
        const int w  = i & 255;
        ((uint4*)s_w4)[i] =
            ((const uint4*)(w4p + (size_t)(kc * 64 + ch * 4) * 512))[w];
      }
      for (int i = tid; i < 512; i += 128)        // 8KB = 512 * uint4
        ((uint4*)s_w5)[i] = ((const uint4*)(w5p + (size_t)ch * 8 * 512))[i];
      // Prefetch next chunk's weights into cache (global_prefetch_b8).
      if (ch + 1 < 16) {
#pragma unroll
        for (int kc = 0; kc < 4; ++kc)
          __builtin_prefetch(w4p + (size_t)(kc * 64 + (ch + 1) * 4) * 512, 0, 0);
        __builtin_prefetch(w5p + (size_t)(ch + 1) * 8 * 512, 0, 0);
      }
      __syncthreads();  // RAW: staged weights visible to all waves

      // Layer 4 chunk: relu(g1@W4+b4) -> per-wave stage (16x64 f16)
#pragma unroll
      for (int nt = 0; nt < 4; ++nt) {
        const int ntg = ch * 4 + nt;
        v8f acc = {};
#pragma unroll
        for (int kc = 0; kc < 4; ++kc)
          acc = wmma16(a4[kc], load_bfrag(s_w4 + (kc * 4 + nt) * 512, lane), acc);
        const float bn = b4[ntg * 16 + col];
#pragma unroll
        for (int rr = 0; rr < 8; ++rr) {
          float v = acc[rr] + bn;
          v = v > 0.0f ? v : 0.0f;
          stage[(rr + mOff) * 64 + nt * 16 + col] = (_Float16)v;
        }
      }
      lds_fence();
      // Layer 5 partial: g3 += g2chunk @ W5[chunk rows]
#pragma unroll
      for (int kc2 = 0; kc2 < 2; ++kc2) {
        const v16h a5 = lds_afrag(stage, 64, kc2 * 32, lane);
#pragma unroll
        for (int nt = 0; nt < 4; ++nt)
          g3[nt] = wmma16(a5, load_bfrag(s_w5 + (kc2 * 4 + nt) * 512, lane), g3[nt]);
      }
    }

    // g3 + b5, relu -> stage (A repack for fc)
#pragma unroll
    for (int nt = 0; nt < 4; ++nt) {
      const float bn = b5[nt * 16 + col];
#pragma unroll
      for (int rr = 0; rr < 8; ++rr) {
        float v = g3[nt][rr] + bn;
        v = v > 0.0f ? v : 0.0f;
        stage[(rr + mOff) * 64 + nt * 16 + col] = (_Float16)v;
      }
    }
    lds_fence();
    const v16h af0 = lds_afrag(stage, 64, 0, lane);
    const v16h af1 = lds_afrag(stage, 64, 32, lane);

    // fc: 16x48 (cols 40..47 zero-padded), stage result in LDS f32.
#pragma unroll
    for (int nt = 0; nt < 3; ++nt) {
      v8f acc = {};
      acc = wmma16(af0, load_bfrag(wfp + (0 * 3 + nt) * 512, lane), acc);
      acc = wmma16(af1, load_bfrag(wfp + (1 * 3 + nt) * 512, lane), acc);
      const int n = nt * 16 + col;
      const float bn = (n < 40) ? bf[n] : 0.0f;
#pragma unroll
      for (int rr = 0; rr < 8; ++rr) outT[(rr + mOff) * 48 + n] = acc[rr] + bn;
    }
    lds_fence();

    // Row-wise log_softmax over 40 classes; one lane per row.
    if (tvalid && lane < 16) {
      const int rg = base + lane;
      if (rg < N_) {
        const float* rp = outT + lane * 48;
        float mx = -3.402823466e38f;
        for (int c = 0; c < 40; ++c) mx = fmaxf(mx, rp[c]);
        float s = 0.0f;
        for (int c = 0; c < 40; ++c) s += __expf(rp[c] - mx);
        const float lse = mx + __logf(s);
        float* op = out + (size_t)rg * 40;
        for (int c = 0; c < 40; ++c) op[c] = rp[c] - lse;
      }
    }
  }
}

extern "C" void kernel_launch(void* const* d_in, const int* in_sizes, int n_in,
                              void* d_out, int out_size, void* d_ws, size_t ws_size,
                              hipStream_t stream) {
  (void)n_in; (void)out_size; (void)ws_size;
  const float*     x   = (const float*)d_in[0];
  const float*     pos = (const float*)d_in[1];
  const long long* ei  = (const long long*)d_in[2];
  const float* W1 = (const float*)d_in[3];
  const float* b1 = (const float*)d_in[4];
  const float* W2 = (const float*)d_in[5];
  const float* b2 = (const float*)d_in[6];
  const float* W3 = (const float*)d_in[7];
  const float* b3 = (const float*)d_in[8];
  const float* W4 = (const float*)d_in[9];
  const float* b4 = (const float*)d_in[10];
  const float* W5 = (const float*)d_in[11];
  const float* b5 = (const float*)d_in[12];
  const float* Wf = (const float*)d_in[13];
  const float* bf = (const float*)d_in[14];
  float* out = (float*)d_out;

  const int Nn = in_sizes[0] / 3;   // 50000
  const int Ee = in_sizes[2] / 2;   // 1600000

  // Workspace carve: agg (N x 64 f32) + packed B-fragment weights.
  char* ws = (char*)d_ws;
  size_t off = 0;
  auto carve = [&](size_t bytes) -> void* {
    void* p = (void*)(ws + off);
    off += bytes;
    off = (off + 255) & ~(size_t)255;
    return p;
  };
  float*     agg = (float*)carve((size_t)Nn * 64 * sizeof(float));
  _Float16* w1p = (_Float16*)carve((size_t)1  * 4  * 512 * 2);
  _Float16* w2p = (_Float16*)carve((size_t)2  * 4  * 512 * 2);
  _Float16* w3p = (_Float16*)carve((size_t)2  * 8  * 512 * 2);
  _Float16* w4p = (_Float16*)carve((size_t)4  * 64 * 512 * 2);
  _Float16* w5p = (_Float16*)carve((size_t)32 * 4  * 512 * 2);
  _Float16* wfp = (_Float16*)carve((size_t)2  * 3  * 512 * 2);

  init_agg<<<2048, 256, 0, stream>>>(agg, Nn * 64);
  pack_b<<<64,  256, 0, stream>>>(W1, w1p, 6,    64,   1,  4);
  pack_b<<<64,  256, 0, stream>>>(W2, w2p, 64,   64,   2,  4);
  pack_b<<<64,  256, 0, stream>>>(W3, w3p, 64,   128,  2,  8);
  pack_b<<<512, 256, 0, stream>>>(W4, w4p, 128,  1024, 4,  64);
  pack_b<<<256, 256, 0, stream>>>(W5, w5p, 1024, 64,   32, 4);
  pack_b<<<16,  256, 0, stream>>>(Wf, wfp, 64,   40,   2,  3);

  const int tilesE = (Ee + Nn + 15) / 16;  // 103125 (exact)
  edge_msg_agg<<<1024, 256, 0, stream>>>(x, pos, ei, b1, b2, w1p, w2p, agg,
                                         Ee, Nn, tilesE);
  const int tilesN = (Nn + 15) / 16;       // 3125 (exact)
  node_mlp<<<512, 128, 0, stream>>>(agg, b3, b4, b5, bf, w3p, w4p, w5p, wfp,
                                    out, Nn, tilesN);
}
